// SpatialTransformerLayer_65953517797885
// MI455X (gfx1250) — matrix operand order repaired and checked by reading, not compiled, not run on hardware
//
#include <hip/hip_runtime.h>

// Problem constants (from reference): B=32, H=W=256, C=32, HID=128
#define B_N      32
#define HW_N     65536            // 256*256
#define C_N      32
#define HID_N    128
#define IMG_N    (HW_N * C_N)     // 2097152 floats per image
#define OUT_ELEMS (B_N * HW_N * C_N) // 67108864 floats (then 192 theta floats)

typedef __attribute__((ext_vector_type(16))) _Float16 v16h;
typedef __attribute__((ext_vector_type(8)))  float    v8f;
typedef __attribute__((ext_vector_type(4)))  float    v4f;

// ---------------------------------------------------------------------------
// Kernel 0: zero the pooling accumulator (d_ws is poisoned, must self-init)
// ---------------------------------------------------------------------------
__global__ __launch_bounds__(256) void stn_zero_acc(float* __restrict__ acc) {
    int t = threadIdx.x;
#pragma unroll
    for (int i = 0; i < 4; ++i) acc[t * 4 + i] = 0.0f;   // 1024 floats
}

// ---------------------------------------------------------------------------
// Kernel 1: global average pool partial sums.
// grid = (64 chunks, 32 images), block = 256 threads.
// Each block reduces a 128KB stripe with 128-bit loads. Because the thread
// stride (1024 floats) is a multiple of C=32, each lane accumulates a fixed
// set of 4 channels -> 4 scalar accumulators, then LDS ds_add_f32 reduce,
// then one global f32 atomic per channel per block.
// ---------------------------------------------------------------------------
__global__ __launch_bounds__(256) void stn_pool(const float* __restrict__ x,
                                                float* __restrict__ acc) {
    const int chunk = blockIdx.x;      // 0..63
    const int b     = blockIdx.y;      // 0..31
    const int t     = threadIdx.x;     // 0..255

    const v4f* p = (const v4f*)(x + (size_t)b * IMG_N + (size_t)chunk * 32768);
    float a0 = 0.f, a1 = 0.f, a2 = 0.f, a3 = 0.f;
#pragma unroll 8
    for (int i = 0; i < 32; ++i) {
        v4f v = p[t + i * 256];
        a0 += v.x; a1 += v.y; a2 += v.z; a3 += v.w;
    }

    __shared__ float sred[C_N];
    if (t < C_N) sred[t] = 0.0f;
    __syncthreads();
    const int c0 = (t * 4) & 31;       // channels c0..c0+3 (no wrap: c0 <= 28)
    atomicAdd(&sred[c0 + 0], a0);
    atomicAdd(&sred[c0 + 1], a1);
    atomicAdd(&sred[c0 + 2], a2);
    atomicAdd(&sred[c0 + 3], a3);
    __syncthreads();
    if (t < C_N) atomicAdd(&acc[b * C_N + t], sred[t]);
}

// ---------------------------------------------------------------------------
// Kernel 2: localization net. One block, 256 threads = 8 waves.
//   feat[32,128] = relu( (acc/65536)[32,32] @ W_loc[32,128] + b_loc )
// via v_wmma_f32_16x16x32_f16: 16 output tiles (2 M-tiles x 8 N-tiles),
// K=32 covered by a single WMMA per tile; 8 waves x 2 tiles each.
//   theta[32,6] = feat @ W_out + b_out   (scalar, 192 dot products)
// Fragment packing follows ISA 7.12.2:
//   A 16x32 f16 : lane half selects K-offset 0/8 (v0..3) resp 16/24 (v4..7)
//   B 32x16 f16 : lane = column, VGPR v holds the same K pair as A (k0, k0+1)
//   D 16x16 f32 : VGPR r -> row = half*8 + r, col = lane&15
// ---------------------------------------------------------------------------
__global__ __launch_bounds__(256) void stn_locnet(const float* __restrict__ acc,
                                                  const float* __restrict__ W_loc,
                                                  const float* __restrict__ b_loc,
                                                  const float* __restrict__ W_out,
                                                  const float* __restrict__ b_out,
                                                  float* __restrict__ theta_ws,
                                                  float* __restrict__ theta_out) {
    __shared__ _Float16 sA[B_N * C_N];       // 2 KB  (A: [batch][channel])
    __shared__ _Float16 sB[C_N * HID_N];     // 8 KB  (B: [channel][hid])
    __shared__ float    sbl[HID_N];          // 0.5 KB
    __shared__ float    sFeat[B_N * HID_N];  // 16 KB

    const int t = threadIdx.x;
    const float inv = 1.0f / 65536.0f;
    for (int i = t; i < B_N * C_N; i += 256)   sA[i] = (_Float16)(acc[i] * inv);
    for (int i = t; i < C_N * HID_N; i += 256) sB[i] = (_Float16)W_loc[i];
    if (t < HID_N) sbl[t] = b_loc[t];
    __syncthreads();

    const int wid  = t >> 5;
    const int lane = t & 31;
    const int mrow = lane & 15;       // row within tile (A) / column (B,D)
    const int half = lane >> 4;       // lane half selects K/M sub-block

#pragma unroll
    for (int i = 0; i < 2; ++i) {
        const int tile = wid * 2 + i;          // 0..15
        const int mt = tile >> 3;              // 0..1
        const int nt = tile & 7;               // 0..7
        v16h afrag, bfrag;
#pragma unroll
        for (int v = 0; v < 8; ++v) {
            const int k0 = (v < 4) ? (half * 8 + 2 * v)
                                   : (16 + half * 8 + 2 * (v - 4));
            afrag[2 * v]     = sA[(mt * 16 + mrow) * C_N + k0];
            afrag[2 * v + 1] = sA[(mt * 16 + mrow) * C_N + k0 + 1];
            const int kb = k0;
            bfrag[2 * v]     = sB[(kb)     * HID_N + nt * 16 + mrow];
            bfrag[2 * v + 1] = sB[(kb + 1) * HID_N + nt * 16 + mrow];
        }
        v8f cacc = {};
        cacc = __builtin_amdgcn_wmma_f32_16x16x32_f16(
            /*neg_a=*/false, afrag, /*neg_b=*/false, bfrag,
            /*c_mod=*/(short)0, cacc, /*reuse_a=*/false, /*reuse_b=*/false);
#pragma unroll
        for (int r = 0; r < 8; ++r) {
            const int row = mt * 16 + half * 8 + r;     // batch index
            const int col = nt * 16 + mrow;             // hidden index
            float v2 = cacc[r] + sbl[col];
            sFeat[row * HID_N + col] = v2 > 0.0f ? v2 : 0.0f;
        }
    }
    __syncthreads();

    if (t < B_N * 6) {
        const int bb = t / 6, j = t % 6;
        float s = b_out[j];
#pragma unroll 8
        for (int h = 0; h < HID_N; ++h) s += sFeat[bb * HID_N + h] * W_out[h * 6 + j];
        theta_ws[t]  = s;
        theta_out[t] = s;   // second element of the output tuple
    }
}

// ---------------------------------------------------------------------------
// Kernel 3: affine grid + bilinear sample. 8 threads per output pixel,
// one 4-channel v4f each -> fully coalesced 128B wave transactions.
// grid = 65536 blocks x 256 threads (32 pixels/block, all same batch image).
// Grid coords computed as (2u-255)/255 so the +-1 endpoints are exact
// (matches linspace endpoint semantics; weights collapse exactly at edges).
// ---------------------------------------------------------------------------
__global__ __launch_bounds__(256) void stn_sample(const float* __restrict__ x,
                                                  const float* __restrict__ theta,
                                                  float* __restrict__ out) {
    __shared__ float th[6];
    const int t    = threadIdx.x;
    const int gid0 = blockIdx.x << 8;
    const int b    = gid0 >> 19;               // pixel0 >> 16
    if (t < 6) th[t] = theta[b * 6 + t];
    __syncthreads();

    const int gid   = gid0 + t;
    const int pixel = gid >> 3;                // 0 .. 32*65536-1
    const int cg    = gid & 7;                 // channel group (4 channels)
    const int p     = pixel & 65535;
    const int uy    = p >> 8;
    const int ux    = p & 255;

    const float gy = (float)(2 * uy - 255) * (1.0f / 255.0f);
    const float gx = (float)(2 * ux - 255) * (1.0f / 255.0f);
    const float ys = (th[0] * gy + th[1] * gx + th[2] + 1.0f) * 127.5f;
    const float xs = (th[3] * gy + th[4] * gx + th[5] + 1.0f) * 127.5f;

    int y0 = (int)floorf(ys); y0 = y0 < 0 ? 0 : (y0 > 255 ? 255 : y0);
    int x0 = (int)floorf(xs); x0 = x0 < 0 ? 0 : (x0 > 255 ? 255 : x0);
    const int y1 = (y0 + 1 > 255) ? 255 : y0 + 1;
    const int x1 = (x0 + 1 > 255) ? 255 : x0 + 1;

    const float y0f = (float)y0, y1f = (float)y1;
    const float x0f = (float)x0, x1f = (float)x1;
    const float w0 = (y1f - ys) * (x1f - xs);
    const float w1 = (y1f - ys) * (xs - x0f);
    const float w2 = (ys - y0f) * (x1f - xs);
    const float w3 = (ys - y0f) * (xs - x0f);

    const v4f* __restrict__ x4 = (const v4f*)x;
    const int rowbase = b << 16;
    const v4f v0 = x4[(rowbase + (y0 << 8) + x0) * 8 + cg];
    const v4f v1 = x4[(rowbase + (y0 << 8) + x1) * 8 + cg];
    const v4f v2 = x4[(rowbase + (y1 << 8) + x0) * 8 + cg];
    const v4f v3 = x4[(rowbase + (y1 << 8) + x1) * 8 + cg];

    const v4f r = w0 * v0 + w1 * v1 + w2 * v2 + w3 * v3;

    // Write-once output: non-temporal store keeps x resident in L2 for reuse.
    __builtin_nontemporal_store(r, &((v4f*)out)[pixel * 8 + cg]);
}

// ---------------------------------------------------------------------------
extern "C" void kernel_launch(void* const* d_in, const int* in_sizes, int n_in,
                              void* d_out, int out_size, void* d_ws, size_t ws_size,
                              hipStream_t stream) {
    const float* x     = (const float*)d_in[0];   // [32,256,256,32]
    const float* W_loc = (const float*)d_in[1];   // [32,128]
    const float* b_loc = (const float*)d_in[2];   // [128]
    const float* W_out = (const float*)d_in[3];   // [128,6]
    const float* b_out = (const float*)d_in[4];   // [6]
    float* out = (float*)d_out;                   // out [32,256,256,32] ++ theta [32,2,3]

    float* acc   = (float*)d_ws;                  // 1024 floats: per-(b,c) sums
    float* theta = acc + 1024;                    // 192 floats

    stn_zero_acc<<<1, 256, 0, stream>>>(acc);
    stn_pool<<<dim3(64, 32), 256, 0, stream>>>(x, acc);
    stn_locnet<<<1, 256, 0, stream>>>(acc, W_loc, b_loc, W_out, b_out,
                                      theta, out + OUT_ELEMS);
    stn_sample<<<65536, 256, 0, stream>>>(x, theta, out);
}
